// GraphConvModel_31035433680973
// MI455X (gfx1250) — matrix operand order repaired and checked by reading, not compiled, not run on hardware
//
#include <hip/hip_runtime.h>
#include <math.h>

// ---------------- problem constants ----------------
#define NN   10000      // nodes  (625 * 16 exactly)
#define EE   160000     // edges
#define HH   512        // feature dim (D == H == 512)
#define CC   10         // classes
#define CPAD 16         // final-conv output columns padded to one WMMA tile
#define LL   3          // hidden conv layers
#define BN_EPS 1e-5f
#define LDSPAD 8        // row padding (halves) -> bank-conflict-free ds loads

typedef __attribute__((ext_vector_type(16))) _Float16 v16h;
typedef __attribute__((ext_vector_type(8)))  _Float16 v8h;
typedef __attribute__((ext_vector_type(8)))  float    v8f;

// ---------------- utility kernels ----------------
__global__ void zero4_kernel(float* __restrict__ p, size_t n4) {
    size_t i = (size_t)blockIdx.x * blockDim.x + threadIdx.x;
    if (i < n4) {
        float4 z = make_float4(0.f, 0.f, 0.f, 0.f);
        reinterpret_cast<float4*>(p)[i] = z;
    }
}

__global__ void degree_kernel(const int* __restrict__ dst, float* __restrict__ deg, int e) {
    int i = blockIdx.x * blockDim.x + threadIdx.x;
    if (i < e) atomicAdd(&deg[dst[i]], 1.0f);
}

__global__ void invdeg_kernel(float* __restrict__ deg, int n) {
    int i = blockIdx.x * blockDim.x + threadIdx.x;
    if (i < n) deg[i] = 1.0f / fmaxf(deg[i], 1.0f);
}

// f32 [rows x HH] -> f16 [padrows x HH], zero-padding rows >= rows.
__global__ void cvt_w_kernel(const float* __restrict__ src, _Float16* __restrict__ dst,
                             int rows, int padrows) {
    int i = blockIdx.x * blockDim.x + threadIdx.x;
    if (i < padrows * HH) {
        int r = i >> 9;                    // HH == 512
        dst[i] = (r < rows) ? (_Float16)src[i] : (_Float16)0.f;
    }
}

// One wave handles one (edge, 128-feature chunk). invdeg[dst] folded in here.
__global__ void scatter_kernel(const float* __restrict__ h,
                               const int* __restrict__ src,
                               const int* __restrict__ dst,
                               const float* __restrict__ invdeg,
                               float* __restrict__ agg) {
    int wave = (blockIdx.x * (blockDim.x >> 5)) + (threadIdx.x >> 5);
    int lane = threadIdx.x & 31;
    if (wave >= EE * 4) return;
    int e = wave >> 2;
    int chunk = wave & 3;
    int s = src[e], d = dst[e];
    float w = invdeg[d];
    const float4 v = *reinterpret_cast<const float4*>(h + (size_t)s * HH + chunk * 128 + lane * 4);
    float* ap = agg + (size_t)d * HH + chunk * 128 + lane * 4;
    atomicAdd(ap + 0, v.x * w);
    atomicAdd(ap + 1, v.y * w);
    atomicAdd(ap + 2, v.z * w);
    atomicAdd(ap + 3, v.w * w);
}

// ---------------- WMMA dual GEMM ----------------
// out[n, j] = sum_k agg[n,k]*Wr[j,k] + sum_k h[n,k]*Ww[j,k]   (agg pre-scaled by invdeg)

// A frag (16-bit A 16x32) from LDS f16 row: lane row m=L&15, g=L>>4;
// elems 0..15 -> K=k0+g*16+e: two 16B ds chunks.
__device__ __forceinline__ v16h load_a_lds(const _Float16* __restrict__ row, int k0, int g) {
    v8h lo = *reinterpret_cast<const v8h*>(row + k0 + (g << 4));
    v8h hi = *reinterpret_cast<const v8h*>(row + k0 + 8 + (g << 4));
    v16h r;
#pragma unroll
    for (int i = 0; i < 8; ++i) { r[i] = lo[i]; r[8 + i] = hi[i]; }
    return r;
}

// B frag (16-bit B 32x16): lane col n=L&15, g=L>>4; elems 0..15 -> K=k0+g*16+e:
// one contiguous 32B f16 chunk per lane (weights pre-converted/padded to f16).
__device__ __forceinline__ v16h load_b_frag(const _Float16* __restrict__ W16, int j, int k0, int g) {
    const v8h* p = reinterpret_cast<const v8h*>(W16 + (size_t)j * HH + k0 + (g << 4));
    v8h lo = p[0], hi = p[1];
    v16h r;
#pragma unroll
    for (int i = 0; i < 8; ++i) { r[i] = lo[i]; r[8 + i] = hi[i]; }
    return r;
}

__global__ void gemm_dual_kernel(const float* __restrict__ Aagg,
                                 const float* __restrict__ Ah,
                                 const _Float16* __restrict__ Wr16,
                                 const _Float16* __restrict__ Ww16,
                                 float* __restrict__ out,
                                 int Hout, int ldout, int do_relu) {
    __shared__ __align__(16) _Float16 shA[2][16][HH + LDSPAD];

    const int tid = threadIdx.x;
    const int lane = tid & 31;
    const int wave = tid >> 5;
    const int wavesPerBlock = blockDim.x >> 5;
    const int n0 = blockIdx.x * 16;
    const int jtile = blockIdx.y * wavesPerBlock + wave;
    const int m = lane & 15;
    const int g = lane >> 4;
    const int j = jtile * 16 + m;            // weight row / output column (padded -> valid)

    __builtin_prefetch(Wr16 + (size_t)j * HH, 0, 1);   // global_prefetch_b8
    __builtin_prefetch(Ww16 + (size_t)j * HH, 0, 1);

    // ---- cooperative stage: both A tiles (agg | h), f32 -> f16, into LDS ----
    // 2 matrices * 16 rows * 64 chunks of 8 elements
    for (int idx = tid; idx < 2 * 16 * (HH / 8); idx += blockDim.x) {
        int which = (idx >= 16 * (HH / 8)) ? 1 : 0;
        int rem = idx - which * 16 * (HH / 8);
        int row = rem >> 6;                  // HH/8 == 64 chunks per row
        int c8 = (rem & 63) << 3;
        const float* srcp = (which ? Ah : Aagg) + (size_t)(n0 + row) * HH + c8;
        float4 v0 = reinterpret_cast<const float4*>(srcp)[0];
        float4 v1 = reinterpret_cast<const float4*>(srcp)[1];
        v8h h8;
        h8[0] = (_Float16)v0.x; h8[1] = (_Float16)v0.y;
        h8[2] = (_Float16)v0.z; h8[3] = (_Float16)v0.w;
        h8[4] = (_Float16)v1.x; h8[5] = (_Float16)v1.y;
        h8[6] = (_Float16)v1.z; h8[7] = (_Float16)v1.w;
        *reinterpret_cast<v8h*>(&shA[which][row][c8]) = h8;
    }
    __syncthreads();

    const _Float16* rowAgg = &shA[0][m][0];
    const _Float16* rowH   = &shA[1][m][0];

    // ---- software-pipelined K loop, two independent accumulator chains ----
    v8f c0 = {}, c1 = {};
    v16h a0 = load_a_lds(rowAgg, 0, g);
    v16h b0 = load_b_frag(Wr16, j, 0, g);
    v16h a1 = load_a_lds(rowH, 0, g);
    v16h b1 = load_b_frag(Ww16, j, 0, g);
#pragma unroll
    for (int k0 = 32; k0 <= HH; k0 += 32) {
        v16h na0 = a0, nb0 = b0, na1 = a1, nb1 = b1;
        if (k0 < HH) {                       // compile-time resolved (full unroll)
            na0 = load_a_lds(rowAgg, k0, g);
            nb0 = load_b_frag(Wr16, j, k0, g);
            na1 = load_a_lds(rowH, k0, g);
            nb1 = load_b_frag(Ww16, j, k0, g);
        }
        c0 = __builtin_amdgcn_wmma_f32_16x16x32_f16(false, a0, false, b0, (short)0, c0, false, false);
        c1 = __builtin_amdgcn_wmma_f32_16x16x32_f16(false, a1, false, b1, (short)0, c1, false, false);
        a0 = na0; b0 = nb0; a1 = na1; b1 = nb1;
    }

    // C/D layout: VGPR r, lanes0-15 -> M=r, lanes16-31 -> M=r+8; N = lane&15.
    const int col = jtile * 16 + m;
    if (col < Hout) {
#pragma unroll
        for (int r = 0; r < 8; ++r) {
            int row = n0 + r + g * 8;
            float v = c0[r] + c1[r];
            if (do_relu) v = fmaxf(v, 0.0f);
            out[(size_t)row * ldout + col] = v;
        }
    }
}

// ---------------- BatchNorm over nodes ----------------
__global__ void bn_partial_kernel(const float* __restrict__ z,
                                  float* __restrict__ colsum,
                                  float* __restrict__ colsum2) {
    int row0 = blockIdx.x * 16;
    for (int cc = 0; cc < HH; cc += blockDim.x) {
        int col = cc + threadIdx.x;
        float s = 0.f, s2 = 0.f;
#pragma unroll 4
        for (int r = 0; r < 16; ++r) {
            float v = z[(size_t)(row0 + r) * HH + col];
            s += v; s2 += v * v;
        }
        atomicAdd(&colsum[col], s);
        atomicAdd(&colsum2[col], s2);
    }
}

__global__ void bn_finalize_kernel(const float* __restrict__ colsum,
                                   const float* __restrict__ colsum2,
                                   const float* __restrict__ gamma,
                                   const float* __restrict__ beta,
                                   float* __restrict__ colscale,
                                   float* __restrict__ colshift) {
    int c = blockIdx.x * blockDim.x + threadIdx.x;
    if (c < HH) {
        float invn = 1.0f / (float)NN;
        float mu = colsum[c] * invn;
        float var = colsum2[c] * invn - mu * mu;
        float sc = gamma[c] * rsqrtf(var + BN_EPS);
        colscale[c] = sc;
        colshift[c] = beta[c] - mu * sc;
    }
}

__global__ void bn_apply_kernel(float* __restrict__ z,
                                const float* __restrict__ colscale,
                                const float* __restrict__ colshift) {
    size_t i4 = ((size_t)blockIdx.x * blockDim.x + threadIdx.x);
    if (i4 >= (size_t)NN * HH / 4) return;
    size_t i = i4 * 4;
    int col = (int)(i & (HH - 1));
    float4 v  = *reinterpret_cast<float4*>(z + i);
    float4 sc = *reinterpret_cast<const float4*>(colscale + col);
    float4 sh = *reinterpret_cast<const float4*>(colshift + col);
    v.x = v.x * sc.x + sh.x;
    v.y = v.y * sc.y + sh.y;
    v.z = v.z * sc.z + sh.z;
    v.w = v.w * sc.w + sh.w;
    *reinterpret_cast<float4*>(z + i) = v;
}

// ---------------- log-softmax (in place, C=10) ----------------
__global__ void logsoftmax_kernel(float* __restrict__ out) {
    int r = blockIdx.x * blockDim.x + threadIdx.x;
    if (r >= NN) return;
    float* p = out + (size_t)r * CC;
    float m = p[0];
#pragma unroll
    for (int c = 1; c < CC; ++c) m = fmaxf(m, p[c]);
    float s = 0.f;
#pragma unroll
    for (int c = 0; c < CC; ++c) s += __expf(p[c] - m);
    float lse = __logf(s) + m;
#pragma unroll
    for (int c = 0; c < CC; ++c) p[c] = p[c] - lse;
}

// ---------------- host launcher ----------------
extern "C" void kernel_launch(void* const* d_in, const int* in_sizes, int n_in,
                              void* d_out, int out_size, void* d_ws, size_t ws_size,
                              hipStream_t stream) {
    (void)in_sizes; (void)n_in; (void)out_size; (void)ws_size;

    const float* x     = (const float*)d_in[0];
    const int*   ei    = (const int*)d_in[1];
    const int*   src   = ei;
    const int*   dst   = ei + EE;
    const float* Wpr   = (const float*)d_in[2];
    const float* Wpw   = (const float*)d_in[3];
    const float* Wrel  = (const float*)d_in[4];   // L x H x H
    const float* Wroot = (const float*)d_in[5];
    const float* gamma = (const float*)d_in[6];   // (L+1) x H
    const float* beta  = (const float*)d_in[7];
    const float* Wfr   = (const float*)d_in[8];   // C x H
    const float* Wfw   = (const float*)d_in[9];
    float* out = (float*)d_out;

    // workspace carve-out
    char* ws = (char*)d_ws;
    size_t off = 0;
    auto carve = [&](size_t bytes) -> void* {
        void* p = (void*)(ws + off);
        off = (off + bytes + 255) & ~(size_t)255;
        return p;
    };
    float* agg      = (float*)carve((size_t)NN * HH * 4);
    float* bufA     = (float*)carve((size_t)NN * HH * 4);
    float* bufB     = (float*)carve((size_t)NN * HH * 4);
    float* deg      = (float*)carve((size_t)NN * 4);     // becomes invdeg in place
    float* colsum   = (float*)carve(2 * HH * 4);         // colsum | colsum2 contiguous
    float* colsum2  = colsum + HH;
    float* colscale = (float*)carve(HH * 4);
    float* colshift = (float*)carve(HH * 4);
    // f16 weight staging (final-conv weights zero-padded to 16 rows)
    _Float16* Wpr16   = (_Float16*)carve((size_t)HH * HH * 2);
    _Float16* Wpw16   = (_Float16*)carve((size_t)HH * HH * 2);
    _Float16* Wrel16  = (_Float16*)carve((size_t)LL * HH * HH * 2);
    _Float16* Wroot16 = (_Float16*)carve((size_t)LL * HH * HH * 2);
    _Float16* Wfr16   = (_Float16*)carve((size_t)CPAD * HH * 2);
    _Float16* Wfw16   = (_Float16*)carve((size_t)CPAD * HH * 2);

    const size_t featN4 = (size_t)NN * HH / 4;
    const int ZB = 256;
    const int WELT = HH * HH;                 // elements in one HxH weight

    // ---- one-time per call: degrees -> invdeg, weights -> f16 ----
    zero4_kernel<<<(int)((NN / 4 + ZB - 1) / ZB), ZB, 0, stream>>>(deg, NN / 4);
    degree_kernel<<<(EE + ZB - 1) / ZB, ZB, 0, stream>>>(dst, deg, EE);
    invdeg_kernel<<<(NN + ZB - 1) / ZB, ZB, 0, stream>>>(deg, NN);

    cvt_w_kernel<<<(WELT + ZB - 1) / ZB, ZB, 0, stream>>>(Wpr, Wpr16, HH, HH);
    cvt_w_kernel<<<(WELT + ZB - 1) / ZB, ZB, 0, stream>>>(Wpw, Wpw16, HH, HH);
    for (int l = 0; l < LL; ++l) {
        cvt_w_kernel<<<(WELT + ZB - 1) / ZB, ZB, 0, stream>>>(
            Wrel + (size_t)l * WELT, Wrel16 + (size_t)l * WELT, HH, HH);
        cvt_w_kernel<<<(WELT + ZB - 1) / ZB, ZB, 0, stream>>>(
            Wroot + (size_t)l * WELT, Wroot16 + (size_t)l * WELT, HH, HH);
    }
    cvt_w_kernel<<<(CPAD * HH + ZB - 1) / ZB, ZB, 0, stream>>>(Wfr, Wfr16, CC, CPAD);
    cvt_w_kernel<<<(CPAD * HH + ZB - 1) / ZB, ZB, 0, stream>>>(Wfw, Wfw16, CC, CPAD);

    auto run_conv = [&](const float* hin, const _Float16* Wr16, const _Float16* Ww16,
                        float* dstbuf, int Hout, int ldout, int relu) {
        zero4_kernel<<<(int)((featN4 + ZB - 1) / ZB), ZB, 0, stream>>>(agg, featN4);
        scatter_kernel<<<EE * 4 / 8, 256, 0, stream>>>(hin, src, dst, deg, agg);
        if (Hout == HH) {
            gemm_dual_kernel<<<dim3(NN / 16, HH / 16 / 8), 256, 0, stream>>>(
                agg, hin, Wr16, Ww16, dstbuf, Hout, ldout, relu);
        } else {
            gemm_dual_kernel<<<dim3(NN / 16, 1), 32, 0, stream>>>(
                agg, hin, Wr16, Ww16, dstbuf, Hout, ldout, relu);
        }
    };
    auto run_bn = [&](float* z, const float* g, const float* b) {
        zero4_kernel<<<(2 * HH / 4 + ZB - 1) / ZB, ZB, 0, stream>>>(colsum, 2 * HH / 4);
        bn_partial_kernel<<<NN / 16, 256, 0, stream>>>(z, colsum, colsum2);
        bn_finalize_kernel<<<(HH + ZB - 1) / ZB, ZB, 0, stream>>>(colsum, colsum2, g, b,
                                                                  colscale, colshift);
        bn_apply_kernel<<<(int)((featN4 + ZB - 1) / ZB), ZB, 0, stream>>>(z, colscale, colshift);
    };

    // proj layer: x -> bufA, relu, bn(gamma[0],beta[0])
    run_conv(x, Wpr16, Wpw16, bufA, HH, HH, 1);
    run_bn(bufA, gamma + 0 * HH, beta + 0 * HH);

    // hidden layers
    float* cur = bufA;
    float* nxt = bufB;
    for (int l = 0; l < LL; ++l) {
        run_conv(cur, Wrel16 + (size_t)l * WELT, Wroot16 + (size_t)l * WELT, nxt, HH, HH, 1);
        run_bn(nxt, gamma + (size_t)(l + 1) * HH, beta + (size_t)(l + 1) * HH);
        float* t = cur; cur = nxt; nxt = t;
    }

    // final conv (C=10, weights padded to 16 cols) straight into d_out, then log-softmax
    run_conv(cur, Wfr16, Wfw16, out, CC, CC, 0);
    logsoftmax_kernel<<<(NN + ZB - 1) / ZB, ZB, 0, stream>>>(out);
}
